// FullAttention_28484223107485
// MI455X (gfx1250) — matrix operand order repaired and checked by reading, not compiled, not run on hardware
//
#include <hip/hip_runtime.h>

#define FDIM 1024
#define HDIM 1024
#define L1DIM 1024
#define L2DIM 1024
#define BATCH 8
#define NLVL 4
#define DLVL 256
#define BIG_NEG (-3.0e38f)

typedef __attribute__((ext_vector_type(16))) __bf16 v16bf;
typedef __attribute__((ext_vector_type(8)))  __bf16 v8bf;
typedef __attribute__((ext_vector_type(8)))  float  v8f;

struct U32x8 { uint4 lo, hi; };

__device__ inline unsigned short bf16b(float f) {
  return __builtin_bit_cast(unsigned short, (__bf16)f);
}

__device__ inline v16bf make_frag(uint4 lo, uint4 hi) {
  U32x8 u;
  u.lo = lo; u.hi = hi;
  return __builtin_bit_cast(v16bf, u);
}

// A-matrix fragment (16x32 bf16): per-lane K = koffA+0..7 and 16+koffA+0..7.
__device__ inline v16bf fragA_bf16(const unsigned short* __restrict__ p) {
  return make_frag(*(const uint4*)p, *(const uint4*)(p + 16));
}

// B-matrix fragment (32x16 bf16): per-lane K = koffB+0..7 and 16+koffB+0..7.
__device__ inline v16bf fragB_bf16(const unsigned short* __restrict__ p) {
  return make_frag(*(const uint4*)p, *(const uint4*)(p + 16));
}

__device__ inline v8f wmma_bf16(v16bf a, v16bf b, v8f c) {
  return __builtin_amdgcn_wmma_f32_16x16x32_bf16(
      false, a, false, b, (short)0, c, false, false);
}

// ---------------------------------------------------------------------------
// Phase 0: bulk fp32 -> bf16 conversion (bandwidth pass).
// ---------------------------------------------------------------------------
__global__ __launch_bounds__(256) void cvt_kernel(const float* __restrict__ src,
                                                  unsigned short* __restrict__ dst,
                                                  int n8) {
  const int idx = blockIdx.x * 256 + threadIdx.x;
  if (idx >= n8) return;
  const float4* p = (const float4*)(src + (size_t)idx * 8);
  float4 a = p[0], b = p[1];
  v8bf r;
  r[0] = (__bf16)a.x; r[1] = (__bf16)a.y; r[2] = (__bf16)a.z; r[3] = (__bf16)a.w;
  r[4] = (__bf16)b.x; r[5] = (__bf16)b.y; r[6] = (__bf16)b.z; r[7] = (__bf16)b.w;
  *(v8bf*)(dst + (size_t)idx * 8) = r;
}

// ---------------------------------------------------------------------------
// Phase 1: keys = relu(Xbf @ Wbf^T) [* final_v] -> bf16 workspace.
// grid.z: 0 -> x1 -> K1, 1 -> x2 -> K2 (scaled by final_v).
// 8 waves (2 M x 4 N), wave tile 64x64 (4x4 WMMA tiles) -> WG tile 128x256.
// Inner loop: 16 b128 loads feeding 16 v_wmma, zero VALU converts.
// ---------------------------------------------------------------------------
__global__ __launch_bounds__(256) void proj_kernel(
    const unsigned short* __restrict__ X1b, const unsigned short* __restrict__ X2b,
    const unsigned short* __restrict__ Wb,  const float* __restrict__ fv,
    unsigned short* __restrict__ K1, unsigned short* __restrict__ K2) {
  const int z = blockIdx.z;
  const unsigned short* __restrict__ X = z ? X2b : X1b;
  unsigned short* __restrict__ O = z ? K2 : K1;

  const int tid = threadIdx.x, lane = tid & 31, wave = tid >> 5;
  const int r15 = lane & 15, half = lane >> 4;
  const int koffA = half * 8, koffB = half * 8;
  const int m0 = blockIdx.x * 128 + (wave >> 2) * 64;  // row tile (B*L)
  const int n0 = blockIdx.y * 256 + (wave & 3) * 64;   // col tile (H)

  v8f acc[4][4] = {};
  const unsigned short* ap[4];
  const unsigned short* bp[4];
#pragma unroll
  for (int i = 0; i < 4; i++)
    ap[i] = X + (size_t)(m0 + i * 16 + r15) * FDIM + koffA;
#pragma unroll
  for (int j = 0; j < 4; j++)
    bp[j] = Wb + (size_t)(n0 + j * 16 + r15) * FDIM + koffB;

  for (int k0 = 0; k0 < FDIM; k0 += 32) {
    v16bf af[4], bfr[4];
#pragma unroll
    for (int i = 0; i < 4; i++) af[i] = fragA_bf16(ap[i] + k0);
#pragma unroll
    for (int j = 0; j < 4; j++) bfr[j] = fragB_bf16(bp[j] + k0);
#pragma unroll
    for (int i = 0; i < 4; i++)
#pragma unroll
      for (int j = 0; j < 4; j++)
        acc[i][j] = wmma_bf16(af[i], bfr[j], acc[i][j]);
  }

#pragma unroll
  for (int i = 0; i < 4; i++) {
#pragma unroll
    for (int j = 0; j < 4; j++) {
      const int col = n0 + j * 16 + r15;
      const float s = z ? fv[col] : 1.0f;
#pragma unroll
      for (int v = 0; v < 8; v++) {
        const int row = m0 + i * 16 + v + half * 8;
        float val = fmaxf(acc[i][j][v], 0.0f) * s;
        O[(size_t)row * HDIM + col] = bf16b(val);
      }
    }
  }
}

// ---------------------------------------------------------------------------
// Phase 2: flash attention per (b, level, 64-row i-block).
// 4 waves x 16 query rows; j loop in steps of 32; O (16x256 f32) resident.
// ---------------------------------------------------------------------------
__global__ __launch_bounds__(128) void attn_kernel(
    const unsigned short* __restrict__ x1k, const unsigned short* __restrict__ x2k,
    const unsigned short* __restrict__ x2b, const unsigned char* __restrict__ mask,
    float* __restrict__ out) {
  __shared__ unsigned short ldsV[DLVL * 32];   // V chunk transposed [d][j], 16KB
  __shared__ unsigned short ldsP[4][16 * 32];  // per-wave P tile, 4KB

  const int b = blockIdx.z, m = blockIdx.y;
  const int tid = threadIdx.x, lane = tid & 31, wave = tid >> 5;
  const int r15 = lane & 15, half = lane >> 4;
  const int koffA = half * 8, koffB = half * 8;
  const int i0 = blockIdx.x * 64 + wave * 16;

  const unsigned short* Q  = x1k + ((size_t)b * L1DIM) * HDIM + m * DLVL;
  const unsigned short* Kp = x2k + ((size_t)b * L2DIM) * HDIM + m * DLVL;
  const unsigned short* Vb = x2b + ((size_t)b * L2DIM) * HDIM + m * DLVL;
  const unsigned char* mb = mask + b * L2DIM;

  v8f Oacc[16];
#pragma unroll
  for (int t = 0; t < 16; t++) Oacc[t] = (v8f){0,0,0,0,0,0,0,0};
  float mrun[8], lrun[8], corr[8];
#pragma unroll
  for (int v = 0; v < 8; v++) { mrun[v] = BIG_NEG; lrun[v] = 0.0f; }

  const unsigned short* qrow = Q + (size_t)(i0 + r15) * HDIM + koffA;

  for (int j0 = 0; j0 < L2DIM; j0 += 32) {
    __syncthreads();
    // Cooperatively stage bf16 V[j0..j0+31][0..255] transposed into LDS.
#pragma unroll
    for (int o = 0; o < 8; o++) {
      const int item = tid + o * 128;
      const int j = item & 31, d0 = (item >> 5) * 8;
      union { uint4 u; unsigned short s[8]; } w;
      w.u = *(const uint4*)(Vb + (size_t)(j0 + j) * HDIM + d0);
#pragma unroll
      for (int e = 0; e < 8; e++) ldsV[(d0 + e) * 32 + j] = w.s[e];
    }
    __syncthreads();

    // S = Q . K^T for two 16-column tiles, K-dim = 256 (8 WMMA steps each).
    v8f S0 = (v8f){0,0,0,0,0,0,0,0}, S1 = (v8f){0,0,0,0,0,0,0,0};
    const unsigned short* k0row = Kp + (size_t)(j0 + r15) * HDIM + koffB;
    const unsigned short* k1row = Kp + (size_t)(j0 + 16 + r15) * HDIM + koffB;
#pragma unroll
    for (int kk = 0; kk < 8; kk++) {
      const int dd = kk * 32;
      v16bf qa  = fragA_bf16(qrow + dd);
      v16bf kb0 = fragB_bf16(k0row + dd);
      v16bf kb1 = fragB_bf16(k1row + dd);
      S0 = wmma_bf16(qa, kb0, S0);
      S1 = wmma_bf16(qa, kb1, S1);
    }

    const bool mk0 = mb[j0 + r15] != 0;
    const bool mk1 = mb[j0 + 16 + r15] != 0;

    // Online softmax (rows live in VGPR index v; columns across 16 lanes).
#pragma unroll
    for (int v = 0; v < 8; v++) {
      float s0 = mk0 ? BIG_NEG : S0[v];
      float s1 = mk1 ? BIG_NEG : S1[v];
      float rmax = fmaxf(s0, s1);
#pragma unroll
      for (int sft = 8; sft >= 1; sft >>= 1)
        rmax = fmaxf(rmax, __shfl_xor(rmax, sft, 32));
      const float mn = fmaxf(mrun[v], rmax);
      const float c  = __expf(mrun[v] - mn);
      const float p0 = __expf(s0 - mn);
      const float p1 = __expf(s1 - mn);
      float rs = p0 + p1;
#pragma unroll
      for (int sft = 8; sft >= 1; sft >>= 1)
        rs += __shfl_xor(rs, sft, 32);
      lrun[v] = lrun[v] * c + rs;
      mrun[v] = mn;
      corr[v] = c;
      // Re-shape P (C layout) -> LDS row-major tile for A-fragment reload.
      ldsP[wave][(v + half * 8) * 32 + r15]      = bf16b(p0);
      ldsP[wave][(v + half * 8) * 32 + 16 + r15] = bf16b(p1);
    }

#pragma unroll
    for (int t = 0; t < 16; t++) {
#pragma unroll
      for (int v = 0; v < 8; v++) Oacc[t][v] *= corr[v];
    }

    asm volatile("s_wait_dscnt 0" ::: "memory");  // wave-local LDS RAW
    v16bf pf = fragA_bf16(&ldsP[wave][r15 * 32 + koffA]);
#pragma unroll
    for (int t = 0; t < 16; t++) {
      v16bf vb = fragB_bf16(&ldsV[(t * 16 + r15) * 32 + koffB]);
      Oacc[t] = wmma_bf16(pf, vb, Oacc[t]);
    }
  }

  float inv[8];
#pragma unroll
  for (int v = 0; v < 8; v++) inv[v] = 1.0f / lrun[v];
#pragma unroll
  for (int t = 0; t < 16; t++) {
#pragma unroll
    for (int v = 0; v < 8; v++) {
      const int row = i0 + v + half * 8;
      const int col = m * DLVL + t * 16 + r15;
      out[((size_t)b * L1DIM + row) * HDIM + col] = Oacc[t][v] * inv[v];
    }
  }
}

extern "C" void kernel_launch(void* const* d_in, const int* in_sizes, int n_in,
                              void* d_out, int out_size, void* d_ws, size_t ws_size,
                              hipStream_t stream) {
  const float* x1_att = (const float*)d_in[0];
  const float* x2_att = (const float*)d_in[1];
  const float* x2     = (const float*)d_in[2];
  const unsigned char* x2_mask = (const unsigned char*)d_in[3];
  const float* W      = (const float*)d_in[4];
  const float* fv     = (const float*)d_in[5];
  float* out = (float*)d_out;

  const size_t NXL = (size_t)BATCH * L1DIM * HDIM;  // 8M elements
  const size_t NW  = (size_t)HDIM * FDIM;           // 1M elements

  unsigned short* x1b = (unsigned short*)d_ws;   // [B*L1, F] bf16
  unsigned short* x2b = x1b + NXL;               // [B*L2, F] bf16
  unsigned short* wb  = x2b + NXL;               // [H, F]    bf16
  unsigned short* vb  = wb + NW;                 // [B*L2, H] bf16 (values)
  unsigned short* k1  = vb + NXL;                // [B*L1, H] bf16 keys
  unsigned short* k2  = k1 + NXL;                // [B*L2, H] bf16 keys

  const int n8x = (int)(NXL / 8), n8w = (int)(NW / 8);
  cvt_kernel<<<(n8x + 255) / 256, 256, 0, stream>>>(x1_att, x1b, n8x);
  cvt_kernel<<<(n8x + 255) / 256, 256, 0, stream>>>(x2_att, x2b, n8x);
  cvt_kernel<<<(n8w + 255) / 256, 256, 0, stream>>>(W, wb, n8w);
  cvt_kernel<<<(n8x + 255) / 256, 256, 0, stream>>>(x2, vb, n8x);

  dim3 g1((BATCH * L1DIM) / 128, HDIM / 256, 2);
  proj_kernel<<<g1, 256, 0, stream>>>(x1b, x2b, wb, fv, k1, k2);

  dim3 g2(L1DIM / 64, NLVL, BATCH);
  attn_kernel<<<g2, 128, 0, stream>>>(k1, k2, vb, x2_mask, out);
}
